// GoEModel_74199855006286
// MI455X (gfx1250) — matrix operand order repaired and checked by reading, not compiled, not run on hardware
//
#include <hip/hip_runtime.h>
#include <hip/hip_bf16.h>

typedef __attribute__((ext_vector_type(16))) _Float16 v16h;
typedef __attribute__((ext_vector_type(8)))  float    v8f;

#define CB 16
#define CS 128
#define CE 512
#define CNE 8
#define CNH 8
#define CFF 2048
#define CRH 256
#define CV 32000
#define CDH 64

// ---------------------------------------------------------------------------
// init: zero final buffer, counts, ent accumulator; set active=1
// ---------------------------------------------------------------------------
__global__ void k_init(float* finalBuf, int* counts, int* active, float* ent, long nFinal) {
    long i0 = (long)blockIdx.x * blockDim.x + threadIdx.x;
    for (long i = i0; i < nFinal; i += (long)gridDim.x * blockDim.x) finalBuf[i] = 0.f;
    if (blockIdx.x == 0) {
        if (threadIdx.x < CB) active[threadIdx.x] = 1;
        if (threadIdx.x < CB * CNE) counts[threadIdx.x] = 0;
        if (threadIdx.x == 0) ent[0] = 0.f;
    }
}

// ---------------------------------------------------------------------------
// generic fp32 -> f16 convert
// ---------------------------------------------------------------------------
__global__ void k_f32_to_f16(const float* __restrict__ src, _Float16* __restrict__ dst, long n) {
    long i0 = (long)blockIdx.x * blockDim.x + threadIdx.x;
    for (long i = i0; i < n; i += (long)gridDim.x * blockDim.x)
        dst[i] = (_Float16)src[i];
}

// ---------------------------------------------------------------------------
// embedding gather * sqrt(E) + sinusoidal positional encoding
// grid = B*S blocks, 256 threads
// ---------------------------------------------------------------------------
__global__ void k_embed(const int* __restrict__ ids, const float* __restrict__ emb,
                        float* __restrict__ x, _Float16* __restrict__ xh) {
    int bs = blockIdx.x;
    int s  = bs & (CS - 1);
    int id = ids[bs];
    const float scale = 22.62741699796952f; // sqrt(512)
    const float kfac  = -9.210340371976184f / (float)CE; // -ln(10000)/E
    for (int e = threadIdx.x; e < CE; e += blockDim.x) {
        int i2 = e >> 1;
        float div = __expf((float)(2 * i2) * kfac);
        float ang = (float)s * div;
        float pe  = (e & 1) ? __cosf(ang) : __sinf(ang);
        float v = emb[(long)id * CE + e] * scale + pe;
        x[(long)bs * CE + e]  = v;
        xh[(long)bs * CE + e] = (_Float16)v;
    }
}

// ---------------------------------------------------------------------------
// router: mean over S, 2-layer MLP, masked softmax, entropy, argmax routing
// single block, 256 threads
// ---------------------------------------------------------------------------
__global__ void k_router(const float* __restrict__ x,
                         const float* __restrict__ r1w, const float* __restrict__ r1b,
                         const float* __restrict__ r2w, const float* __restrict__ r2b,
                         int* counts, int* active, int* route, int* term, int* eidx,
                         float* entAcc) {
    __shared__ float summ[CB][CE];
    __shared__ float hh[CB][CRH];
    __shared__ float entv[CB];
    __shared__ int   oldact[CB];
    int t = threadIdx.x; // 256

    // summ[b][e] = mean over s of x[b,s,e]
    for (int b = 0; b < CB; b++) {
        for (int e = t; e < CE; e += 256) {
            const float* px = x + ((long)b * CS) * CE + e;
            float acc = 0.f;
            for (int s = 0; s < CS; s++) acc += px[(long)s * CE];
            summ[b][e] = acc * (1.f / (float)CS);
        }
    }
    __syncthreads();

    // hh[b][r] = relu(summ[b,:] . r1w[r,:] + r1b[r]),  r = t
    {
        float acc[CB];
        #pragma unroll
        for (int b = 0; b < CB; b++) acc[b] = 0.f;
        const float* w = r1w + (long)t * CE;
        for (int k = 0; k < CE; k++) {
            float wv = w[k];
            #pragma unroll
            for (int b = 0; b < CB; b++) acc[b] += wv * summ[b][k];
        }
        float bv = r1b[t];
        for (int b = 0; b < CB; b++) hh[b][t] = fmaxf(acc[b] + bv, 0.f);
    }
    __syncthreads();

    if (t < CB) {
        int b = t;
        float l[CNE + 1];
        for (int o = 0; o <= CNE; o++) {
            float acc = r2b[o];
            const float* w = r2w + (long)o * CRH;
            for (int r = 0; r < CRH; r++) acc += hh[b][r] * w[r];
            l[o] = acc;
        }
        for (int e = 0; e < CNE; e++)
            if (counts[b * CNE + e] >= 2) l[e] = -__builtin_inff();
        float mx = l[0];
        for (int o = 1; o <= CNE; o++) mx = fmaxf(mx, l[o]);
        float p[CNE + 1], ssum = 0.f;
        for (int o = 0; o <= CNE; o++) { p[o] = __expf(l[o] - mx); ssum += p[o]; }
        float ent = 0.f;
        for (int o = 0; o <= CNE; o++) {
            float pi = p[o] / ssum;
            ent -= pi * __logf(pi + 1e-9f);
        }
        int dec = 0; float best = l[0];
        for (int o = 1; o <= CNE; o++) if (l[o] > best) { best = l[o]; dec = o; }
        int act = active[b];
        oldact[b] = act;
        entv[b] = act ? ent : 0.f;
        int tm = (act && dec == CNE) ? 1 : 0;
        int rt = (act && dec <  CNE) ? 1 : 0;
        term[b] = tm;
        route[b] = rt;
        int ei = rt ? dec : 0;
        eidx[b] = ei;
        if (rt) counts[b * CNE + ei] += 1;
        active[b] = rt;
    }
    __syncthreads();
    if (t == 0) {
        int nact = 0; float se = 0.f;
        for (int b = 0; b < CB; b++) { nact += oldact[b]; se += entv[b]; }
        entAcc[0] += se / (float)(nact > 0 ? nact : 1);
    }
}

// ---------------------------------------------------------------------------
// conditional row copy: if flag[b], final[b,s,:] = x[b,s,:]
// ---------------------------------------------------------------------------
__global__ void k_select_final(const float* __restrict__ x, float* __restrict__ fin,
                               const int* __restrict__ flag) {
    int bs = blockIdx.x;
    int b = bs >> 7;
    if (!flag[b]) return;
    for (int e = threadIdx.x; e < CE; e += blockDim.x)
        fin[(long)bs * CE + e] = x[(long)bs * CE + e];
}

// ---------------------------------------------------------------------------
// row softmax over 128 columns, fp32 in -> f16 out. grid = B*NH*S, 128 thr
// ---------------------------------------------------------------------------
__global__ void k_softmax(const float* __restrict__ sc, _Float16* __restrict__ ph) {
    long row = blockIdx.x;
    int t = threadIdx.x;
    __shared__ float red[128];
    float v = sc[row * 128 + t];
    red[t] = v; __syncthreads();
    for (int st = 64; st > 0; st >>= 1) { if (t < st) red[t] = fmaxf(red[t], red[t + st]); __syncthreads(); }
    float mx = red[0];
    __syncthreads();
    float ev = __expf(v - mx);
    red[t] = ev; __syncthreads();
    for (int st = 64; st > 0; st >>= 1) { if (t < st) red[t] += red[t + st]; __syncthreads(); }
    float s = red[0];
    ph[row * 128 + t] = (_Float16)(ev / s);
}

// ---------------------------------------------------------------------------
// fused residual + LayerNorm (+ optional tag add, optional conditional commit)
// ---------------------------------------------------------------------------
__global__ void k_ln(const float* __restrict__ resid, const float* __restrict__ delta,
                     const float* __restrict__ g, const float* __restrict__ beta,
                     const float* __restrict__ tag, const int* __restrict__ eidx,
                     const int* __restrict__ commit,
                     float* __restrict__ outF, _Float16* __restrict__ outH) {
    int bs = blockIdx.x;
    int b = bs >> 7;
    if (commit && !commit[b]) return;
    int e = eidx[b];
    __shared__ float rs[256], rq[256];
    int t = threadIdx.x;
    const float* r0 = resid + (long)bs * CE;
    const float* d0 = delta + (long)bs * CE;
    float v0 = r0[t]       + d0[t];
    float v1 = r0[t + 256] + d0[t + 256];
    rs[t] = v0 + v1; rq[t] = v0 * v0 + v1 * v1;
    __syncthreads();
    for (int st = 128; st > 0; st >>= 1) { if (t < st) { rs[t] += rs[t + st]; rq[t] += rq[t + st]; } __syncthreads(); }
    float mean = rs[0] * (1.f / (float)CE);
    float var  = rq[0] * (1.f / (float)CE) - mean * mean;
    float inv  = rsqrtf(var + 1e-5f);
    const float* gp = g    + (long)e * CE;
    const float* bp = beta + (long)e * CE;
    float t0 = tag ? tag[(long)e * CE + t]       : 0.f;
    float t1 = tag ? tag[(long)e * CE + t + 256] : 0.f;
    float o0 = (v0 - mean) * inv * gp[t]       + bp[t]       + t0;
    float o1 = (v1 - mean) * inv * gp[t + 256] + bp[t + 256] + t1;
    outF[(long)bs * CE + t]       = o0;
    outF[(long)bs * CE + t + 256] = o1;
    outH[(long)bs * CE + t]       = (_Float16)o0;
    outH[(long)bs * CE + t + 256] = (_Float16)o1;
}

__global__ void k_ent_out(const float* ent, float* out) { out[0] = ent[0]; }

// ---------------------------------------------------------------------------
// Batched WMMA GEMM with async-to-LDS A staging.
//   Block = 4 waves covering a 16(M) x 64(N) strip; all waves share one A
//   tile per K-step. The 16x32 f16 A tile (1KB) is staged into LDS by
//   GLOBAL_LOAD_ASYNC_TO_LDS_B64 (8B per lane x 128 threads), double
//   buffered: each wave issues its portion, waits its own ASYNCcnt
//   (s_wait_asynccnt 0), then the workgroup barrier publishes the tile.
//   B fragments stream directly from global (b128 + prefetch) - their reuse
//   lives in L2 (all f16 weights fit in the 192MB L2).
//   C[m,n] = scale * sum_k A[m,k] * (TRANSB ? B[n,k] : B[k,n]) + bias[n]
// Fragment layouts: contiguous k order, lane half h holds K 16h..16h+15 in A and B (16-bit A 16x32, B 32x16).
// ---------------------------------------------------------------------------
template <bool TRANSB, bool RELU, bool EIDX>
__global__ void k_gemm(const _Float16* __restrict__ A, long aOuter, long aInner,
                       const _Float16* __restrict__ Bm, long bOuter, long bInner,
                       float* __restrict__ C, long cOuter, long cInner,
                       const float* __restrict__ bias, long biasStride,
                       const int* __restrict__ eidx,
                       int inner, int Mtiles, int Ntiles4,
                       int K, int lda, int ldb, int ldc, float scale) {
    __shared__ _Float16 shA[2][16 * 32];

    int w    = threadIdx.x >> 5;  // wave in block: selects N tile
    int lane = threadIdx.x & 31;
    int tilesPer = Mtiles * Ntiles4;
    int bb  = blockIdx.x / tilesPer;
    int tt  = blockIdx.x - bb * tilesPer;
    int mt  = tt / Ntiles4;
    int nt  = (tt - mt * Ntiles4) * 4 + w;

    const _Float16* Ab = A + (long)(bb / inner) * aOuter + (long)(bb % inner) * aInner;
    const _Float16* Bb;
    const float* bp = nullptr;
    if (EIDX) {
        int ex = eidx[bb];
        Bb = Bm + (long)ex * bOuter;
        if (bias) bp = bias + (long)ex * biasStride;
    } else {
        Bb = Bm + (long)(bb / inner) * bOuter + (long)(bb % inner) * bInner;
        bp = bias;
    }
    float* Cb = C + (long)(bb / inner) * cOuter + (long)(bb % inner) * cInner;

    int m0 = mt * 16, n0 = nt * 16;
    int half = lane >> 4;      // 0: lanes 0-15, 1: lanes 16-31
    int lm   = lane & 15;
    int kbA  = half ? 16 : 0;  // A: lanes<16 hold K 0..15; lanes>=16 hold K 16..31
    int kbB  = half ? 16 : 0;  // B: lanes<16 hold K 0..15; lanes>=16 hold K 16..31

    // cooperative A staging: thread -> (row, 4-half chunk) of the 16x32 tile
    int arow = threadIdx.x >> 3;        // 0..15
    int acol = (threadIdx.x & 7) * 4;   // halfs, 8 bytes per thread
    const _Float16* gA = Ab + (long)(m0 + arow) * lda + acol;

    auto issueA = [&](int buf, int k0) {
        unsigned ldsAddr = (unsigned)(size_t)&shA[buf][arow * 32 + acol];
        unsigned long long gsrc = (unsigned long long)(size_t)(gA + k0);
        asm volatile("global_load_async_to_lds_b64 %0, %1, off"
                     :: "v"(ldsAddr), "v"(gsrc) : "memory");
    };

    const _Float16* brow = TRANSB ? (Bb + (long)(n0 + lm) * ldb) : nullptr;

    issueA(0, 0); // prologue

    v8f acc = {};
    int buf = 0;
    for (int k0 = 0; k0 < K; k0 += 32, buf ^= 1) {
        asm volatile("s_wait_asynccnt 0x0" ::: "memory");
        __syncthreads();                       // A tile for this K-step visible
        if (k0 + 32 < K) issueA(buf ^ 1, k0 + 32);

        union { v16h v; uint4 u[2]; _Float16 h[16]; } af, bf;
        const uint4* sa = (const uint4*)&shA[buf][lm * 32 + kbA];
        af.u[0] = sa[0];   // K = kbA + 0..7
        af.u[1] = sa[1];   // K = kbA + 8..15
        if (TRANSB) {
            const uint4* pb = (const uint4*)(brow + k0 + kbB);
            bf.u[0] = pb[0];
            bf.u[1] = pb[1];
            if (k0 + 32 < K)
                __builtin_prefetch(brow + k0 + 32 + kbB, 0, 1);
        } else {
            const _Float16* bcol = Bb + n0 + lm;
            #pragma unroll
            for (int e = 0; e < 16; e++)
                bf.h[e] = bcol[(long)(k0 + kbB + e) * ldb];
        }
        acc = __builtin_amdgcn_wmma_f32_16x16x32_f16(
            false, af.v, false, bf.v, (short)0, acc, false, false);
        __syncthreads();                       // all reads done before rewrite
    }

    float bv = bp ? bp[n0 + lm] : 0.f;
    #pragma unroll
    for (int v = 0; v < 8; v++) {
        int m = m0 + v + half * 8;
        float val = acc[v] * scale + bv;
        if (RELU) val = fmaxf(val, 0.f);
        Cb[(long)m * ldc + n0 + lm] = val;
    }
}

// ---------------------------------------------------------------------------
extern "C" void kernel_launch(void* const* d_in, const int* in_sizes, int n_in,
                              void* d_out, int out_size, void* d_ws, size_t ws_size,
                              hipStream_t stream) {
    const int*   ids   = (const int*)  d_in[0];
    const float* emb   = (const float*)d_in[1];
    const float* in_w  = (const float*)d_in[2];
    const float* in_b  = (const float*)d_in[3];
    const float* out_w = (const float*)d_in[4];
    const float* out_b = (const float*)d_in[5];
    const float* ln1_g = (const float*)d_in[6];
    const float* ln1_b = (const float*)d_in[7];
    const float* ln2_g = (const float*)d_in[8];
    const float* ln2_b = (const float*)d_in[9];
    const float* ff1_w = (const float*)d_in[10];
    const float* ff1_b = (const float*)d_in[11];
    const float* ff2_w = (const float*)d_in[12];
    const float* ff2_b = (const float*)d_in[13];
    const float* tag   = (const float*)d_in[14];
    const float* r1_w  = (const float*)d_in[15];
    const float* r1_b  = (const float*)d_in[16];
    const float* r2_w  = (const float*)d_in[17];
    const float* r2_b  = (const float*)d_in[18];
    const float* lm_w  = (const float*)d_in[19];
    const float* lm_b  = (const float*)d_in[20];
    float* out = (float*)d_out;

    // ---- workspace layout -------------------------------------------------
    char* ws = (char*)d_ws;
    size_t off = 0;
    auto alloc = [&](size_t bytes) -> void* {
        void* p = ws + off;
        off += (bytes + 255) & ~(size_t)255;
        return p;
    };
    const long nInW  = (long)CNE * 3 * CE * CE;
    const long nOutW = (long)CNE * CE * CE;
    const long nFf1W = (long)CNE * CFF * CE;
    const long nFf2W = (long)CNE * CE * CFF;
    const long nLmW  = (long)CV * CE;
    const long nX    = (long)CB * CS * CE;
    const long nQkv  = (long)CB * CS * 3 * CE;
    const long nSc   = (long)CB * CNH * CS * CS;
    const long nH    = (long)CB * CS * CFF;

    _Float16* in_wh  = (_Float16*)alloc(nInW  * 2);
    _Float16* out_wh = (_Float16*)alloc(nOutW * 2);
    _Float16* ff1_wh = (_Float16*)alloc(nFf1W * 2);
    _Float16* ff2_wh = (_Float16*)alloc(nFf2W * 2);
    _Float16* lm_wh  = (_Float16*)alloc(nLmW  * 2);
    float*    x      = (float*)   alloc(nX * 4);
    _Float16* xh     = (_Float16*)alloc(nX * 2);
    float*    finalB = (float*)   alloc(nX * 4);
    _Float16* finalH = (_Float16*)alloc(nX * 2);
    float*    qkv    = (float*)   alloc(nQkv * 4);
    _Float16* qkvh   = (_Float16*)alloc(nQkv * 2);
    float*    scores = (float*)   alloc(nSc * 4);
    _Float16* ph     = (_Float16*)alloc(nSc * 2);
    float*    obuf   = (float*)   alloc(nX * 4);
    _Float16* oh     = (_Float16*)alloc(nX * 2);
    float*    attnO  = (float*)   alloc(nX * 4);
    float*    x1     = (float*)   alloc(nX * 4);
    _Float16* x1h    = (_Float16*)alloc(nX * 2);
    float*    hbuf   = (float*)   alloc(nH * 4);
    _Float16* hh     = (_Float16*)alloc(nH * 2);
    float*    ffout  = (float*)   alloc(nX * 4);
    int*      counts = (int*)     alloc(CB * CNE * 4);
    int*      active = (int*)     alloc(CB * 4);
    int*      route  = (int*)     alloc(CB * 4);
    int*      term   = (int*)     alloc(CB * 4);
    int*      eidx   = (int*)     alloc(CB * 4);
    float*    entAcc = (float*)   alloc(4);
    if (off > ws_size) return;

    const int CT = 256, CG = 8192; // convert kernel config
    // ---- init + weight conversion ----------------------------------------
    k_init<<<4096, 256, 0, stream>>>(finalB, counts, active, entAcc, nX);
    k_f32_to_f16<<<CG, CT, 0, stream>>>(in_w,  in_wh,  nInW);
    k_f32_to_f16<<<CG, CT, 0, stream>>>(out_w, out_wh, nOutW);
    k_f32_to_f16<<<CG, CT, 0, stream>>>(ff1_w, ff1_wh, nFf1W);
    k_f32_to_f16<<<CG, CT, 0, stream>>>(ff2_w, ff2_wh, nFf2W);
    k_f32_to_f16<<<CG, CT, 0, stream>>>(lm_w,  lm_wh,  nLmW);

    k_embed<<<CB * CS, 256, 0, stream>>>(ids, emb, x, xh);

    // ---- ponder loop ------------------------------------------------------
    for (int step = 0; step < 4; step++) {
        k_router<<<1, 256, 0, stream>>>(x, r1_w, r1_b, r2_w, r2_b,
                                        counts, active, route, term, eidx, entAcc);
        k_select_final<<<CB * CS, 256, 0, stream>>>(x, finalB, term);

        // qkv = x @ in_w[e]^T + in_b[e]   [B][128,1536], K=512
        k_gemm<true, false, true><<<CB * 8 * 24, 128, 0, stream>>>(
            xh, (long)CS * CE, 0,
            in_wh, (long)3 * CE * CE, 0,
            qkv, (long)CS * 3 * CE, 0,
            in_b, 3 * CE, eidx,
            1, 8, 24, CE, CE, CE, 3 * CE, 1.f);
        k_f32_to_f16<<<CG, CT, 0, stream>>>(qkv, qkvh, nQkv);

        // scores = q k^T / sqrt(DH)   [B*NH][128,128], K=64
        k_gemm<true, false, false><<<CB * CNH * 8 * 2, 128, 0, stream>>>(
            qkvh, (long)CS * 3 * CE, CDH,
            qkvh + CE, (long)CS * 3 * CE, CDH,
            scores, (long)CNH * CS * CS, (long)CS * CS,
            nullptr, 0, nullptr,
            CNH, 8, 2, CDH, 3 * CE, 3 * CE, CS, 0.125f);
        k_softmax<<<CB * CNH * CS, 128, 0, stream>>>(scores, ph);

        // o = att @ v   [B*NH][128,64], K=128  (NN)
        k_gemm<false, false, false><<<CB * CNH * 8 * 1, 128, 0, stream>>>(
            ph, (long)CNH * CS * CS, (long)CS * CS,
            qkvh + 2 * CE, (long)CS * 3 * CE, CDH,
            obuf, (long)CS * CE, CDH,
            nullptr, 0, nullptr,
            CNH, 8, 1, CS, CS, 3 * CE, CE, 1.f);
        k_f32_to_f16<<<CG, CT, 0, stream>>>(obuf, oh, nX);

        // attn out proj  [B][128,512], K=512
        k_gemm<true, false, true><<<CB * 8 * 8, 128, 0, stream>>>(
            oh, (long)CS * CE, 0,
            out_wh, (long)CE * CE, 0,
            attnO, (long)CS * CE, 0,
            out_b, CE, eidx,
            1, 8, 8, CE, CE, CE, CE, 1.f);
        k_ln<<<CB * CS, 256, 0, stream>>>(x, attnO, ln1_g, ln1_b, nullptr, eidx, nullptr, x1, x1h);

        // ff1 = relu(x1 @ ff1_w[e]^T + b)   [B][128,2048], K=512
        k_gemm<true, true, true><<<CB * 8 * 32, 128, 0, stream>>>(
            x1h, (long)CS * CE, 0,
            ff1_wh, (long)CFF * CE, 0,
            hbuf, (long)CS * CFF, 0,
            ff1_b, CFF, eidx,
            1, 8, 32, CE, CE, CE, CFF, 1.f);
        k_f32_to_f16<<<CG, CT, 0, stream>>>(hbuf, hh, nH);

        // ff2 = h @ ff2_w[e]^T + b   [B][128,512], K=2048
        k_gemm<true, false, true><<<CB * 8 * 8, 128, 0, stream>>>(
            hh, (long)CS * CFF, 0,
            ff2_wh, (long)CE * CFF, 0,
            ffout, (long)CS * CE, 0,
            ff2_b, CE, eidx,
            1, 8, 8, CFF, CFF, CFF, CE, 1.f);
        // LN2 + tag, conditionally committed into x/xh where routed
        k_ln<<<CB * CS, 256, 0, stream>>>(x1, ffout, ln2_g, ln2_b, tag, eidx, route, x, xh);
    }

    // ---- finalize + LM head ----------------------------------------------
    k_select_final<<<CB * CS, 256, 0, stream>>>(x, finalB, active);
    k_f32_to_f16<<<CG, CT, 0, stream>>>(finalB, finalH, nX);

    // lm = final @ lm_w^T + lm_b   [2048, 32000], K=512
    k_gemm<true, false, false><<<(CB * CS / 16) * (CV / 64), 128, 0, stream>>>(
        finalH, 0, 0,
        lm_wh, 0, 0,
        out, 0, 0,
        lm_b, 0, nullptr,
        1, CB * CS / 16, CV / 64, CE, CE, CE, CV, 1.f);
    k_ent_out<<<1, 1, 0, stream>>>(entAcc, out + (long)out_size - 1);
}
